// ResidualBlock_87170656240159
// MI455X (gfx1250) — matrix operand order repaired and checked by reading, not compiled, not run on hardware
//
#include <hip/hip_runtime.h>

// CDNA5 / gfx1250: wave32, WMMA f16 -> f32 accumulate.
typedef __attribute__((ext_vector_type(16))) _Float16 v16h;
typedef __attribute__((ext_vector_type(8)))  _Float16 v8h;
typedef __attribute__((ext_vector_type(8)))  float    v8f;

#define DDIM   512
#define TM     128          // rows per block
#define TN     128          // cols per N pass
#define PAD    8            // halves of LDS row padding (16B) -> bank-conflict-free
#define LROW   (DDIM + PAD) // 520 halves per LDS row
#define NWAVES 8
#define LN_EPS 1e-6f

// ---------------------------------------------------------------------------
// Pre-pack dense kernel (f32, KxN row-major) into WMMA B-fragment order, f16.
// B tile (kb, nb) covers K=[kb*32,kb*32+32) x N=[nb*16,nb*16+16).
// B 32x16, contiguous k order (same as A): lane L holds column (L&15), K range
// (L>>4)*16 .. +16, two halves per VGPR. We store each tile as 32 lanes x 16
// contiguous halves so the GEMM does a single contiguous 32B load per lane.
// ---------------------------------------------------------------------------
__global__ __launch_bounds__(256) void pack_b_f16(const float* __restrict__ kn,
                                                  _Float16* __restrict__ wsB) {
    int tid    = blockIdx.x * 256 + threadIdx.x;   // 0 .. 512*512-1
    int tile   = tid >> 9;                          // (nb*16 + kb)
    int within = tid & 511;
    int lane   = within >> 4;                       // 0..31
    int j      = within & 15;
    int nb     = tile >> 4;
    int kb     = tile & 15;
    int k      = kb * 32 + ((lane >> 4) << 4) + j;  // K index
    int col    = (nb << 4) + (lane & 15);           // N index
    wsB[tid] = (_Float16)kn[k * DDIM + col];
}

// ---------------------------------------------------------------------------
// Fused: LayerNorm + ReLU -> LDS(f16) -> WMMA GEMM -> +x +bias -> out
// ---------------------------------------------------------------------------
__global__ __launch_bounds__(256) void fused_ln_relu_gemm(
    const float* __restrict__ x, const float* __restrict__ ln_scale,
    const float* __restrict__ ln_bias, const _Float16* __restrict__ wsB,
    const float* __restrict__ bias, float* __restrict__ out) {

    __shared__ __align__(32) _Float16 sh[TM * LROW];   // 130 KB (<= 320 KB/WGP)

    const int lane = threadIdx.x & 31;
    const int wave = threadIdx.x >> 5;
    const int m0   = blockIdx.x * TM;

    // ---------------- Phase A: LayerNorm + ReLU into LDS -------------------
    const int cbase = lane << 4;                 // this lane's 16 columns
    float ls[16], lb[16];
    {
        const float4* s4 = (const float4*)(ln_scale + cbase);
        const float4* b4 = (const float4*)(ln_bias + cbase);
        for (int q = 0; q < 4; ++q) {
            float4 a = s4[q];  float4 b = b4[q];
            ls[4*q+0] = a.x; ls[4*q+1] = a.y; ls[4*q+2] = a.z; ls[4*q+3] = a.w;
            lb[4*q+0] = b.x; lb[4*q+1] = b.y; lb[4*q+2] = b.z; lb[4*q+3] = b.w;
        }
    }

    for (int i = 0; i < TM / NWAVES; ++i) {           // 16 rows per wave
        const int rl = wave * (TM / NWAVES) + i;
        const float4* x4 = (const float4*)(x + (size_t)(m0 + rl) * DDIM + cbase);
        float v[16];
        {
            float4 t0 = x4[0], t1 = x4[1], t2 = x4[2], t3 = x4[3];
            v[0]=t0.x;  v[1]=t0.y;  v[2]=t0.z;  v[3]=t0.w;
            v[4]=t1.x;  v[5]=t1.y;  v[6]=t1.z;  v[7]=t1.w;
            v[8]=t2.x;  v[9]=t2.y;  v[10]=t2.z; v[11]=t2.w;
            v[12]=t3.x; v[13]=t3.y; v[14]=t3.z; v[15]=t3.w;
        }
        float s = 0.f, ss = 0.f;
        for (int j = 0; j < 16; ++j) { s += v[j]; ss += v[j] * v[j]; }
        for (int off = 16; off >= 1; off >>= 1) {      // wave32 reduction
            s  += __shfl_xor(s,  off, 32);
            ss += __shfl_xor(ss, off, 32);
        }
        const float mu   = s * (1.0f / DDIM);
        const float var  = ss * (1.0f / DDIM) - mu * mu;
        const float rstd = rsqrtf(var + LN_EPS);

        v8h h0, h1;
        for (int j = 0; j < 8; ++j) {
            h0[j] = (_Float16)fmaxf((v[j]     - mu) * rstd * ls[j]     + lb[j],     0.f);
            h1[j] = (_Float16)fmaxf((v[j + 8] - mu) * rstd * ls[j + 8] + lb[j + 8], 0.f);
        }
        _Float16* dst = &sh[rl * LROW + cbase];
        *(v8h*)dst       = h0;
        *(v8h*)(dst + 8) = h1;
    }
    __syncthreads();

    // ---------------- Phase B: WMMA GEMM over LDS stripe --------------------
    // 4 waves along M (2 tiles each), 2 waves along N (4 tiles each).
    const int wm = wave & 3;
    const int wn = wave >> 2;

    for (int nit = 0; nit < DDIM / TN; ++nit) {
        const int n0 = nit * TN;
        v8f acc[2][4] = {};

        for (int kb = 0; kb < DDIM / 32; ++kb) {
            // A fragments from LDS: contiguous k order (matches packed B):
            // lane L: row = L&15, halves 0..15 = K koff..koff+15,
            // with koff = kb*32 + (L>=16 ? 16 : 0).
            union AU { v16h v; v8h h[2]; } a[2];
            const int koff = kb * 32 + ((lane >> 4) << 4);
            for (int mi = 0; mi < 2; ++mi) {
                const int rl = ((wm * 2 + mi) << 4) + (lane & 15);
                const _Float16* ap = &sh[rl * LROW + koff];
                a[mi].h[0] = *(const v8h*)ap;
                a[mi].h[1] = *(const v8h*)(ap + 8);
            }
            for (int ni = 0; ni < 4; ++ni) {
                const int nbG = (n0 >> 4) + wn * 4 + ni;          // global 16-col block
                const v16h b = *(const v16h*)(wsB + (((size_t)(nbG * 16 + kb)) << 9)
                                                   + (lane << 4));
                for (int mi = 0; mi < 2; ++mi) {
                    acc[mi][ni] = __builtin_amdgcn_wmma_f32_16x16x32_f16(
                        false, a[mi].v, false, b, (short)0, acc[mi][ni],
                        false, false);
                }
            }
        }

        // Epilogue: C layout (ISA 7.12.2): VGPR r holds row r (lanes 0-15)
        // or row r+8 (lanes 16-31); column = lane&15.
        for (int ni = 0; ni < 4; ++ni) {
            const int gcol = n0 + ((wn * 4 + ni) << 4) + (lane & 15);
            const float bv = bias[gcol];
            for (int mi = 0; mi < 2; ++mi) {
                const int grow = m0 + ((wm * 2 + mi) << 4) + ((lane >> 4) << 3);
                v8f c = acc[mi][ni];
                for (int r = 0; r < 8; ++r) {
                    const size_t idx = (size_t)(grow + r) * DDIM + gcol;
                    out[idx] = x[idx] + c[r] + bv;
                }
            }
        }
    }
}

// ---------------------------------------------------------------------------
extern "C" void kernel_launch(void* const* d_in, const int* in_sizes, int n_in,
                              void* d_out, int out_size, void* d_ws, size_t ws_size,
                              hipStream_t stream) {
    const float* x  = (const float*)d_in[0];
    const float* ls = (const float*)d_in[1];
    const float* lb = (const float*)d_in[2];
    const float* kn = (const float*)d_in[3];
    const float* bs = (const float*)d_in[4];
    float* out = (float*)d_out;

    _Float16* wsB = (_Float16*)d_ws;   // 512*512 halves = 512 KB (L2-resident)

    pack_b_f16<<<(DDIM * DDIM) / 256, 256, 0, stream>>>(kn, wsB);

    const int nrows = in_sizes[0] / DDIM;          // 262144
    fused_ln_relu_gemm<<<nrows / TM, 256, 0, stream>>>(x, ls, lb, wsB, bs, out);
}
